// LocalGrouper_12506944766653
// MI455X (gfx1250) — matrix operand order repaired and checked by reading, not compiled, not run on hardware
//
#include <hip/hip_runtime.h>
#include <hip/hip_bf16.h>

typedef __attribute__((ext_vector_type(16))) _Float16 v16h;
typedef __attribute__((ext_vector_type(8)))  _Float16 v8h;
typedef __attribute__((ext_vector_type(8)))  float    v8f;
typedef __attribute__((ext_vector_type(4)))  float    v4f;

// Problem constants
#define BB   8
#define NN   4096
#define SS   1024
#define CC   64
#define KK   32
#define COLS 32768           // SS*KK, col = s*32 + k
#define LDK1 96              // Cin 70 padded to 3*32
#define OC1  128
#define OC2  256
#define NBINS 64
#define CNT_F 262144.0f      // B*K*S

// ---------------------------------------------------------------------------
// Kernel 0: exact top-32 nearest neighbors per query via LDS argmin rounds.
// One wave per query; 2 waves / 64-thread block -> 32KB LDS.
// ---------------------------------------------------------------------------
__global__ __launch_bounds__(64) void topk_kernel(
    const float* __restrict__ xyz, const float* __restrict__ new_xyz,
    int* __restrict__ idx_out) {
  __shared__ float sd[2 * NN];
  const int lane = threadIdx.x & 31;
  const int w    = threadIdx.x >> 5;
  const int q    = blockIdx.x * 2 + w;          // 0..8191
  const int b    = q >> 10;
  const int s    = q & 1023;
  const float qx = new_xyz[(size_t)q * 3 + 0];
  const float qy = new_xyz[(size_t)q * 3 + 1];
  const float qz = new_xyz[(size_t)q * 3 + 2];
  float* d = sd + w * NN;
  const float* xb = xyz + (size_t)b * NN * 3;
  for (int j = 0; j < NN / 32; ++j) {
    const int n = j * 32 + lane;
    const float dx = xb[n * 3 + 0] - qx;
    const float dy = xb[n * 3 + 1] - qy;
    const float dz = xb[n * 3 + 2] - qz;
    d[n] = dx * dx + dy * dy + dz * dz;
  }
  __syncthreads();
  for (int r = 0; r < KK; ++r) {
    float bd = 3.0e38f;
    int   bn = 0;
    for (int j = 0; j < NN / 32; ++j) {
      const int n = j * 32 + lane;
      const float v = d[n];
      if (v < bd) { bd = v; bn = n; }
    }
    // wave32 butterfly argmin (tie-break on smaller index)
    for (int off = 16; off > 0; off >>= 1) {
      const float od = __shfl_xor(bd, off);
      const int   on = __shfl_xor(bn, off);
      if (od < bd || (od == bd && on < bn)) { bd = od; bn = on; }
    }
    if ((bn & 31) == lane) d[bn] = 3.0e38f;   // invalidate winner
    if (lane == 0) idx_out[q * KK + r] = bn;
  }
}

// ---------------------------------------------------------------------------
// Kernel 1: gather + concat -> Xt[b][col][96] f16 (rows 70..95 zero)
// ---------------------------------------------------------------------------
__global__ void gather_kernel(const float* __restrict__ points,
                              const float* __restrict__ xyz,
                              const float* __restrict__ new_xyz,
                              const int* __restrict__ idx,
                              _Float16* __restrict__ Xt) {
  const size_t t = (size_t)blockIdx.x * blockDim.x + threadIdx.x;
  const size_t total = (size_t)BB * COLS * LDK1;
  if (t >= total) return;
  const int i   = (int)(t % LDK1);
  const size_t bc = t / LDK1;
  const int col = (int)(bc % COLS);
  const int b   = (int)(bc / COLS);
  const int s   = col >> 5;
  const int k   = col & 31;
  const int n   = idx[(((size_t)b << 10) + s) * KK + k];
  float v;
  if (i < 64)       v = points[((size_t)b * CC + i) * NN + n];
  else if (i < 67)  v = xyz[((size_t)b * NN + n) * 3 + (i - 64)];
  else if (i < 70)  v = new_xyz[((size_t)b * SS + s) * 3 + (i - 67)];
  else              v = 0.0f;
  Xt[t] = (_Float16)v;
}

// ---------------------------------------------------------------------------
// Weight conversion: W1[128,70]->f16[128,96] zero-padded; W2[256,128]->f16
// ---------------------------------------------------------------------------
__global__ void prep_w_kernel(const float* __restrict__ W1,
                              const float* __restrict__ W2,
                              _Float16* __restrict__ W1h,
                              _Float16* __restrict__ W2h) {
  const int t = blockIdx.x * blockDim.x + threadIdx.x;
  if (t < OC1 * LDK1) {
    const int o = t / LDK1, i = t % LDK1;
    W1h[t] = (_Float16)(i < 70 ? W1[o * 70 + i] : 0.0f);
  } else {
    const int u = t - OC1 * LDK1;
    if (u < OC2 * OC1) W2h[u] = (_Float16)W2[u];
  }
}

__global__ void zero_kernel(float* __restrict__ p, int n) {
  const int t = blockIdx.x * blockDim.x + threadIdx.x;
  if (t < n) p[t] = 0.0f;
}

// ---------------------------------------------------------------------------
// WMMA GEMM: Z[b][o][col] = sum_i W[o][i]*X[b][col][i] + bias[o]
// Wave tile: 16 cols x 64 channels, KSTEPS k-steps of 32.
// Accumulates per-channel sum / sum-of-squares into binned fp32 partials.
// ---------------------------------------------------------------------------
template <int KSTEPS, int OC>
__global__ __launch_bounds__(256) void gemm_kernel(
    const _Float16* __restrict__ Xt,    // [B][COLS][KSTEPS*32]
    const _Float16* __restrict__ Wh,    // [OC][KSTEPS*32]
    const float* __restrict__ bias,     // [OC]
    float* __restrict__ Z,              // [B][OC][COLS]
    float* __restrict__ stats) {        // [NBINS][2][OC]
  constexpr int NT  = 4;                // channel tiles per wave (64 channels)
  constexpr int LDX = KSTEPS * 32;
  const int lane = threadIdx.x & 31;
  const int wid  = (int)((blockIdx.x * blockDim.x + threadIdx.x) >> 5);
  constexpr int chanGroups = OC / (NT * 16);
  constexpr int colTiles   = COLS / 16;
  const int cg = wid % chanGroups;
  const int ct = (wid / chanGroups) % colTiles;
  const int b  = wid / (chanGroups * colTiles);
  if (b >= BB) return;
  const int colBase  = ct * 16;
  const int chanBase = cg * (NT * 16);
  const int hs       = (lane & 16);     // lane-half select

  const _Float16* aptr =
      Xt + ((size_t)b * COLS + colBase + (lane & 15)) * LDX + (hs ? 16 : 0);
  const _Float16* bptr[NT];
#pragma unroll
  for (int t = 0; t < NT; ++t)
    bptr[t] = Wh + (size_t)(chanBase + t * 16 + (lane & 15)) * LDX + (hs ? 16 : 0);

  v8f acc[NT] = {};
  union AV { v16h v; v8h h[2]; };
#pragma unroll
  for (int ks = 0; ks < KSTEPS; ++ks) {
    AV A;
    A.h[0] = *(const v8h*)(aptr + ks * 32);        // K = base .. base+7
    A.h[1] = *(const v8h*)(aptr + ks * 32 + 8);    // K = base+8 .. +15
#pragma unroll
    for (int t = 0; t < NT; ++t) {
      const v16h Bf = *(const v16h*)(bptr[t] + ks * 32);
      acc[t] = __builtin_amdgcn_wmma_f32_16x16x32_f16(
          false, A.v, false, Bf, (short)0, acc[t], false, false);
    }
  }

  const int colOff = colBase + (hs ? 8 : 0);
  float* sbin = stats + (size_t)(wid & (NBINS - 1)) * 2 * OC;
#pragma unroll
  for (int t = 0; t < NT; ++t) {
    const int o  = chanBase + t * 16 + (lane & 15);
    const float bi = bias[o];
    float ls = 0.0f, lq = 0.0f;
    v4f lo, hi;
#pragma unroll
    for (int r = 0; r < 4; ++r) {
      const float v0 = acc[t][r] + bi;
      const float v1 = acc[t][r + 4] + bi;
      lo[r] = v0; hi[r] = v1;
      ls += v0 + v1;
      lq += v0 * v0 + v1 * v1;
    }
    float* zp = Z + ((size_t)b * OC + o) * COLS + colOff;
    *(v4f*)(zp)     = lo;
    *(v4f*)(zp + 4) = hi;
    ls += __shfl_xor(ls, 16);
    lq += __shfl_xor(lq, 16);
    if (lane < 16) {
      atomicAdd(&sbin[o], ls);
      atomicAdd(&sbin[OC + o], lq);
    }
  }
}

// ---------------------------------------------------------------------------
// BN finalize: a[o] = g*rstd ; c[o] = be - mu*g*rstd
// ---------------------------------------------------------------------------
__global__ void finalize_kernel(const float* __restrict__ stats,
                                const float* __restrict__ g,
                                const float* __restrict__ be,
                                float* __restrict__ ac, int OC) {
  const int o = blockIdx.x * blockDim.x + threadIdx.x;
  if (o >= OC) return;
  float s = 0.0f, q = 0.0f;
  for (int bin = 0; bin < NBINS; ++bin) {
    s += stats[(size_t)bin * 2 * OC + o];
    q += stats[(size_t)bin * 2 * OC + OC + o];
  }
  const float mu  = s / CNT_F;
  const float var = q / CNT_F - mu * mu;
  const float a   = g[o] * rsqrtf(var + 1e-5f);
  ac[o]      = a;
  ac[OC + o] = be[o] - mu * a;
}

// BN+ReLU on Z1, transpose to channel-contiguous f16 H1t[b][col][128]
__global__ void bnrelu_kernel(const float* __restrict__ Z1,
                              const float* __restrict__ ac,
                              _Float16* __restrict__ H1t) {
  const size_t t = (size_t)blockIdx.x * blockDim.x + threadIdx.x;
  const size_t total = (size_t)BB * OC1 * COLS;
  if (t >= total) return;
  const int n = (int)(t % COLS);
  const int o = (int)((t / COLS) % OC1);
  const int b = (int)(t / ((size_t)COLS * OC1));
  float v = Z1[t] * ac[o] + ac[OC1 + o];
  v = v > 0.0f ? v : 0.0f;
  H1t[((size_t)b * COLS + n) * OC1 + o] = (_Float16)v;
}

// BN+ReLU on Z2 + max over K=32 (cols s*32..s*32+31) -> new_points
__global__ void maxout_kernel(const float* __restrict__ Z2,
                              const float* __restrict__ ac,
                              float* __restrict__ out_np) {
  const size_t t = (size_t)blockIdx.x * blockDim.x + threadIdx.x;
  const size_t total = (size_t)BB * OC2 * SS;
  if (t >= total) return;
  const int s = (int)(t % SS);
  const int o = (int)((t / SS) % OC2);
  const int b = (int)(t / ((size_t)SS * OC2));
  const float a = ac[o], c = ac[OC2 + o];
  const float* zp = Z2 + ((size_t)b * OC2 + o) * COLS + (size_t)s * KK;
  float m = 0.0f;  // all candidates are ReLU'd (>=0)
#pragma unroll 8
  for (int k = 0; k < KK; ++k) {
    float v = zp[k] * a + c;
    v = v > 0.0f ? v : 0.0f;
    m = m > v ? m : v;
  }
  out_np[((size_t)b * OC2 + o) * SS + s] = m;
}

__global__ void copy_xyz_kernel(const float* __restrict__ src,
                                float* __restrict__ dst, int n) {
  const int t = blockIdx.x * blockDim.x + threadIdx.x;
  if (t < n) dst[t] = src[t];
}

// ---------------------------------------------------------------------------
extern "C" void kernel_launch(void* const* d_in, const int* in_sizes, int n_in,
                              void* d_out, int out_size, void* d_ws,
                              size_t ws_size, hipStream_t stream) {
  const float* xyz     = (const float*)d_in[0];
  const float* points  = (const float*)d_in[1];
  const float* new_xyz = (const float*)d_in[2];
  const float* W1 = (const float*)d_in[3];
  const float* b1 = (const float*)d_in[4];
  const float* g1 = (const float*)d_in[5];
  const float* be1 = (const float*)d_in[6];
  const float* W2 = (const float*)d_in[7];
  const float* b2 = (const float*)d_in[8];
  const float* g2 = (const float*)d_in[9];
  const float* be2 = (const float*)d_in[10];
  float* out = (float*)d_out;

  // workspace carve (256B aligned)
  char* w = (char*)d_ws;
  auto carve = [&](size_t bytes) {
    char* p = w;
    w += (bytes + 255) & ~(size_t)255;
    return p;
  };
  int*      idx  = (int*)carve((size_t)BB * SS * KK * 4);
  _Float16* Xt   = (_Float16*)carve((size_t)BB * COLS * LDK1 * 2);
  _Float16* W1h  = (_Float16*)carve((size_t)OC1 * LDK1 * 2);
  _Float16* W2h  = (_Float16*)carve((size_t)OC2 * OC1 * 2);
  float*    Z1   = (float*)carve((size_t)BB * OC1 * COLS * 4);
  _Float16* H1t  = (_Float16*)carve((size_t)BB * COLS * OC1 * 2);
  float*    Z2   = (float*)carve((size_t)BB * OC2 * COLS * 4);
  float*    st1  = (float*)carve((size_t)NBINS * 2 * OC1 * 4);
  float*    st2  = (float*)carve((size_t)NBINS * 2 * OC2 * 4);
  float*    ac1  = (float*)carve((size_t)2 * OC1 * 4);
  float*    ac2  = (float*)carve((size_t)2 * OC2 * 4);

  // stats must be re-zeroed every call (graph replays don't re-poison)
  {
    const int n1 = NBINS * 2 * OC1, n2 = NBINS * 2 * OC2;
    zero_kernel<<<(n1 + 255) / 256, 256, 0, stream>>>(st1, n1);
    zero_kernel<<<(n2 + 255) / 256, 256, 0, stream>>>(st2, n2);
  }

  // top-32 neighbors
  topk_kernel<<<(BB * SS) / 2, 64, 0, stream>>>(xyz, new_xyz, idx);

  // gather + concat -> f16 Xt
  {
    const size_t total = (size_t)BB * COLS * LDK1;
    gather_kernel<<<(unsigned)((total + 255) / 256), 256, 0, stream>>>(
        points, xyz, new_xyz, idx, Xt);
  }

  // weights -> f16
  {
    const int total = OC1 * LDK1 + OC2 * OC1;
    prep_w_kernel<<<(total + 255) / 256, 256, 0, stream>>>(W1, W2, W1h, W2h);
  }

  // layer 1 GEMM (+ BN stats)
  {
    const int waves = BB * (COLS / 16) * (OC1 / 64);
    gemm_kernel<3, OC1><<<waves * 32 / 256, 256, 0, stream>>>(
        Xt, W1h, b1, Z1, st1);
  }
  finalize_kernel<<<1, OC1, 0, stream>>>(st1, g1, be1, ac1, OC1);
  {
    const size_t total = (size_t)BB * OC1 * COLS;
    bnrelu_kernel<<<(unsigned)((total + 255) / 256), 256, 0, stream>>>(
        Z1, ac1, H1t);
  }

  // layer 2 GEMM (+ BN stats)
  {
    const int waves = BB * (COLS / 16) * (OC2 / 64);
    gemm_kernel<4, OC2><<<waves * 32 / 256, 256, 0, stream>>>(
        H1t, W2h, b2, Z2, st2);
  }
  finalize_kernel<<<1, OC2, 0, stream>>>(st2, g2, be2, ac2, OC2);

  // BN+ReLU + max over K, outputs
  copy_xyz_kernel<<<(BB * SS * 3 + 255) / 256, 256, 0, stream>>>(
      new_xyz, out, BB * SS * 3);
  {
    const size_t total = (size_t)BB * OC2 * SS;
    maxout_kernel<<<(unsigned)((total + 255) / 256), 256, 0, stream>>>(
        Z2, ac2, out + BB * SS * 3);
  }
}
